// MambaSandwich_49941879718201
// MI455X (gfx1250) — compile-verified, matrix operand order checked by reading, not run on hardware
//
#include <hip/hip_runtime.h>
#include <math.h>

// ---------------- model constants ----------------
#define B_SZ    4
#define SEQ     2048
#define DIN     768
#define DM      320
#define DI_     640      // D_MODEL * EXPAND
#define DS      160      // D_STATE
#define DTR     20       // DT_RANK
#define CONVW   4
#define WIN_    8
#define NHEADS  4
#define DH      80       // DM / NHEADS
#define M_ROWS  (B_SZ * SEQ)   // 8192 tokens, multiple of 64

typedef __attribute__((ext_vector_type(16))) _Float16 v16h;
typedef __attribute__((ext_vector_type(8)))  _Float16 v8h;
typedef __attribute__((ext_vector_type(8)))  float    v8f;
typedef __attribute__((ext_vector_type(4)))  float    v4f;

// ---------------------------------------------------------------
// Weight transpose + cast:  W[K][N] f32  ->  Wt[N][K] f16
// ---------------------------------------------------------------
__global__ __launch_bounds__(256) void wcast_kernel(
    const float* __restrict__ W, _Float16* __restrict__ Wt, int K, int N)
{
    int idx = blockIdx.x * 256 + threadIdx.x;
    if (idx >= K * N) return;
    int n = idx / K, k = idx % K;
    Wt[idx] = (_Float16)W[(size_t)k * N + n];
}

// ---------------------------------------------------------------
// WMMA GEMM: C[M x N] = A[M x K] (f32) * W (f16, pre-transposed Wt[N][K])
//            (+ bias[n]) (+ residual[m][n])
// Block = 128 threads = 4 waves; each wave owns a 16(M) x 64(N) tile
// (4 f32 accumulators, A fragment reused 4x). K multiple of 32.
// Out-of-range N columns: W row pointer clamped (branch-free) so the
// K-loop runs with full EXEC and no guards; stores are column-guarded.
// Fragment layouts per ISA 7.12.2:
//   A (16-bit 16x32): lane row M=lane&15; halves 0..7 = K[base0..+7],
//     halves 8..15 = K[base1..+7], base0=kb+8*(lane>>4), base1=base0+16.
//   B mirrors A with N in place of M (contiguous K in Wt row-major).
//   C/D f32 16x16: VGPR r -> M = r + 8*(lane>>4), N = lane&15.
// ---------------------------------------------------------------
__global__ __launch_bounds__(128) void gemm_wmma_kernel(
    const float*    __restrict__ A,
    const _Float16* __restrict__ Wt,
    const float*    __restrict__ bias,    // nullable
    const float*    __restrict__ resid,   // nullable, [M x N]
    float*          __restrict__ C,
    int K, int N)
{
    const int lane = threadIdx.x & 31;
    const int wave = threadIdx.x >> 5;
    const int grp  = lane >> 4;
    const int mr   = lane & 15;
    const int m0   = (blockIdx.x * 4 + wave) * 16;
    const int n0   = blockIdx.y * 64;

    const float* Arow = A + (size_t)(m0 + mr) * K;

    int ncol[4];
    const _Float16* Wrow[4];
#pragma unroll
    for (int j = 0; j < 4; ++j) {
        const int n = n0 + j * 16 + mr;
        ncol[j] = n;
        const int nc = (n < N) ? n : 0;       // clamp: garbage cols never stored
        Wrow[j] = Wt + (size_t)nc * K;
    }

    v8f acc[4] = {};

    for (int kb = 0; kb < K; kb += 32) {
        const int base0 = kb + grp * 8;
        const int base1 = base0 + 16;

        // A fragment: 16 f32 -> 16 f16 (v_cvt_pk_f16_f32)
        v4f a0 = *(const v4f*)(Arow + base0);
        v4f a1 = *(const v4f*)(Arow + base0 + 4);
        v4f a2 = *(const v4f*)(Arow + base1);
        v4f a3 = *(const v4f*)(Arow + base1 + 4);
        v16h af;
#pragma unroll
        for (int i = 0; i < 4; ++i) {
            af[i]      = (_Float16)a0[i];
            af[4 + i]  = (_Float16)a1[i];
            af[8 + i]  = (_Float16)a2[i];
            af[12 + i] = (_Float16)a3[i];
        }

        __builtin_prefetch(Arow + kb + 32, 0, 1);   // speculative global_prefetch_b8

        // 4 B fragments, 4 WMMAs reusing the A fragment
#pragma unroll
        for (int j = 0; j < 4; ++j) {
            v8h b0 = *(const v8h*)(Wrow[j] + base0);
            v8h b1 = *(const v8h*)(Wrow[j] + base1);
            v16h bf;
#pragma unroll
            for (int i = 0; i < 8; ++i) { bf[i] = b0[i]; bf[8 + i] = b1[i]; }
            acc[j] = __builtin_amdgcn_wmma_f32_16x16x32_f16(
                false, af, false, bf, (short)0, acc[j], false, false);
        }
    }

#pragma unroll
    for (int j = 0; j < 4; ++j) {
        const int n = ncol[j];
        if (n < N) {
            const float bv = bias ? bias[n] : 0.0f;
#pragma unroll
            for (int r = 0; r < 8; ++r) {
                const int mm = m0 + r + grp * 8;
                float val = acc[j][r] + bv;
                if (resid) val += resid[(size_t)mm * N + n];
                C[(size_t)mm * N + n] = val;
            }
        }
    }
}

// ---------------------------------------------------------------
// LayerNorm: one wave per row (D = 320), 8 rows per block
// ---------------------------------------------------------------
__global__ __launch_bounds__(256) void layernorm_kernel(
    const float* __restrict__ X, const float* __restrict__ s,
    const float* __restrict__ b, float* __restrict__ Y, int D)
{
    const int row  = blockIdx.x * 8 + (threadIdx.x >> 5);
    const int lane = threadIdx.x & 31;
    const float* x = X + (size_t)row * D;
    float sum = 0.f;
    for (int i = lane; i < D; i += 32) sum += x[i];
    for (int off = 16; off; off >>= 1) sum += __shfl_xor(sum, off, 32);
    const float mean = sum / (float)D;
    float var = 0.f;
    for (int i = lane; i < D; i += 32) { float d = x[i] - mean; var += d * d; }
    for (int off = 16; off; off >>= 1) var += __shfl_xor(var, off, 32);
    const float inv = rsqrtf(var / (float)D + 1e-5f);
    float* y = Y + (size_t)row * D;
    for (int i = lane; i < D; i += 32) y[i] = (x[i] - mean) * inv * s[i] + b[i];
}

// ---------------------------------------------------------------
// Causal depthwise conv (CONV=4) + SiLU.  u = t0[:, 0:640] (stride 1280)
// ---------------------------------------------------------------
__global__ __launch_bounds__(256) void conv_silu_kernel(
    const float* __restrict__ t0, const float* __restrict__ cw,
    const float* __restrict__ cb, float* __restrict__ uc)
{
    const int idx = blockIdx.x * 256 + threadIdx.x;
    if (idx >= M_ROWS * DI_) return;
    const int m = idx / DI_, c = idx % DI_;
    const int l = m % SEQ;
    float acc = cb[c];
#pragma unroll
    for (int i = 0; i < CONVW; ++i) {
        const int lo = l - (CONVW - 1) + i;
        if (lo >= 0)
            acc += t0[(size_t)(m - (CONVW - 1) + i) * (2 * DI_) + c] * cw[c * CONVW + i];
    }
    uc[idx] = acc / (1.0f + __expf(-acc));   // silu
}

// ---------------------------------------------------------------
// delta = softplus(dt @ dt_w + dt_b); dt = xdbl[:, 0:20] (stride 340)
// ---------------------------------------------------------------
__global__ __launch_bounds__(256) void dt_kernel(
    const float* __restrict__ xdbl, const float* __restrict__ dtw,
    const float* __restrict__ dtb, float* __restrict__ delta)
{
    const int idx = blockIdx.x * 256 + threadIdx.x;
    if (idx >= M_ROWS * DI_) return;
    const int m = idx / DI_, d = idx % DI_;
    float acc = dtb[d];
    const float* dr = xdbl + (size_t)m * (DTR + 2 * DS);
#pragma unroll
    for (int r = 0; r < DTR; ++r) acc += dr[r] * dtw[r * DI_ + d];
    delta[idx] = (acc > 20.f) ? acc : log1pf(__expf(acc));
}

// ---------------------------------------------------------------
// Selective scan. Block = 256 threads = 16 d's x 16 state-groups.
// Each thread keeps 10 states + 10 A values in registers; 16-lane
// shuffle reduction produces y. Sequential over L, parallel over
// B x DI x 16 state-groups = 40960 lanes.
// ---------------------------------------------------------------
__global__ __launch_bounds__(256) void scan_kernel(
    const float* __restrict__ delta, const float* __restrict__ uc,
    const float* __restrict__ xdbl,  const float* __restrict__ Alog,
    float* __restrict__ ys)
{
    const int b    = blockIdx.x / (DI_ / 16);
    const int dblk = blockIdx.x % (DI_ / 16);
    const int dloc = threadIdx.x >> 4;
    const int sg   = threadIdx.x & 15;
    const int d    = dblk * 16 + dloc;
    const int s0   = sg * (DS / 16);          // 10 states per thread

    float a[10], h[10];
#pragma unroll
    for (int j = 0; j < 10; ++j) {
        a[j] = -__expf(Alog[(size_t)d * DS + s0 + j]);
        h[j] = 0.f;
    }

    for (int t = 0; t < SEQ; ++t) {
        const size_t m  = (size_t)b * SEQ + t;
        const float dt  = delta[m * DI_ + d];
        const float du  = dt * uc[m * DI_ + d];
        const float* Bp = xdbl + m * (DTR + 2 * DS) + DTR + s0;
        const float* Cp = xdbl + m * (DTR + 2 * DS) + DTR + DS + s0;
        float yp = 0.f;
#pragma unroll
        for (int j = 0; j < 10; ++j) {
            const float hv = h[j] * __expf(dt * a[j]) + du * Bp[j];
            h[j] = hv;
            yp += hv * Cp[j];
        }
        yp += __shfl_xor(yp, 8, 16);
        yp += __shfl_xor(yp, 4, 16);
        yp += __shfl_xor(yp, 2, 16);
        yp += __shfl_xor(yp, 1, 16);
        if (sg == 0) ys[m * DI_ + d] = yp;
    }
}

// ---------------------------------------------------------------
// Gate: out = (ys + uc*Dp) * silu(z);  z = t0[:, 640:1280]
// ---------------------------------------------------------------
__global__ __launch_bounds__(256) void gate_kernel(
    const float* __restrict__ ys, const float* __restrict__ uc,
    const float* __restrict__ Dp, const float* __restrict__ t0,
    float* __restrict__ out)
{
    const int idx = blockIdx.x * 256 + threadIdx.x;
    if (idx >= M_ROWS * DI_) return;
    const int m = idx / DI_, d = idx % DI_;
    const float z = t0[(size_t)m * (2 * DI_) + DI_ + d];
    const float y = ys[idx] + uc[idx] * Dp[d];
    out[idx] = y * (z / (1.0f + __expf(-z)));
}

// ---------------------------------------------------------------
// Local window attention (WIN=8, causal). One wave per (b,l,h).
// dh=80 -> lane dims {lane, lane+32, lane+64(lane<16)}.
// ---------------------------------------------------------------
__global__ __launch_bounds__(256) void attn_kernel(
    const float* __restrict__ q, const float* __restrict__ k,
    const float* __restrict__ v, float* __restrict__ o)
{
    const int w    = blockIdx.x * 8 + (threadIdx.x >> 5);
    const int lane = threadIdx.x & 31;
    const int hh   = w % NHEADS;
    const int l    = (w / NHEADS) % SEQ;
    const int b    = w / (NHEADS * SEQ);
    const size_t rb = ((size_t)b * SEQ + l) * DM + hh * DH;

    const float qd0 = q[rb + lane];
    const float qd1 = q[rb + 32 + lane];
    const float qd2 = (lane < 16) ? q[rb + 64 + lane] : 0.f;

    float sc[WIN_];
#pragma unroll
    for (int ww = 0; ww < WIN_; ++ww) {
        const int ls = l - ww;            // wave-uniform
        float p = -1e9f;
        if (ls >= 0) {
            const size_t kb = ((size_t)b * SEQ + ls) * DM + hh * DH;
            p = qd0 * k[kb + lane] + qd1 * k[kb + 32 + lane];
            if (lane < 16) p += qd2 * k[kb + 64 + lane];
            for (int off = 16; off; off >>= 1) p += __shfl_xor(p, off, 32);
            p *= 0.11180339887498949f;    // 1/sqrt(80)
        }
        sc[ww] = p;
    }

    float mx = sc[0];
#pragma unroll
    for (int ww = 1; ww < WIN_; ++ww) mx = fmaxf(mx, sc[ww]);
    float den = 0.f, pr[WIN_];
#pragma unroll
    for (int ww = 0; ww < WIN_; ++ww) { pr[ww] = __expf(sc[ww] - mx); den += pr[ww]; }
    const float inv = 1.0f / den;

    float o0 = 0.f, o1 = 0.f, o2 = 0.f;
#pragma unroll
    for (int ww = 0; ww < WIN_; ++ww) {
        const int ls = l - ww;
        if (ls >= 0) {
            const size_t vb = ((size_t)b * SEQ + ls) * DM + hh * DH;
            const float aw = pr[ww] * inv;
            o0 += aw * v[vb + lane];
            o1 += aw * v[vb + 32 + lane];
            if (lane < 16) o2 += aw * v[vb + 64 + lane];
        }
    }
    o[rb + lane]      = o0;
    o[rb + 32 + lane] = o1;
    if (lane < 16) o[rb + 64 + lane] = o2;
}

// ---------------------------------------------------------------
// Final L2-normalize of rows of [8192 x 768]
// ---------------------------------------------------------------
__global__ __launch_bounds__(256) void outnorm_kernel(
    const float* __restrict__ X, float* __restrict__ out)
{
    const int row  = blockIdx.x * 8 + (threadIdx.x >> 5);
    const int lane = threadIdx.x & 31;
    const float* x = X + (size_t)row * DIN;
    float ss = 0.f;
    for (int i = lane; i < DIN; i += 32) { const float vv = x[i]; ss += vv * vv; }
    for (int off = 16; off; off >>= 1) ss += __shfl_xor(ss, off, 32);
    const float inv = 1.0f / fmaxf(sqrtf(ss), 1e-12f);
    float* y = out + (size_t)row * DIN;
    for (int i = lane; i < DIN; i += 32) y[i] = x[i] * inv;
}

// ===============================================================
// Host orchestration
// ===============================================================
extern "C" void kernel_launch(void* const* d_in, const int* in_sizes, int n_in,
                              void* d_out, int out_size, void* d_ws, size_t ws_size,
                              hipStream_t stream)
{
    (void)in_sizes; (void)n_in; (void)out_size; (void)ws_size;
    const float* x_in = (const float*)d_in[0];
    auto in = [&](int i) { return (const float*)d_in[i]; };

    // input layout: 0:x, 1:ip.w, 2:ip.b, [3..22] 2 front attn (10 each),
    // [23..110] 8 mamba (11 each), [111..130] 2 back attn, 131: head_w
    const int FRONT0 = 3, TRUNK0 = 23, BACK0 = 111, HEADW = 131;

    // workspace layout (floats)
    float* ws = (float*)d_ws;
    size_t off = 0;
    float* hbuf = ws + off; off += (size_t)M_ROWS * DM;        // activations
    float* hn   = ws + off; off += (size_t)M_ROWS * DM;        // layernorm out
    float* t0   = ws + off; off += (size_t)M_ROWS * 2 * DI_;   // [u|z] or [q|k|v|o]
    float* t1   = ws + off; off += (size_t)M_ROWS * (DTR + 2 * DS); // xdbl
    float* ucb  = ws + off; off += (size_t)M_ROWS * DI_;
    float* dlt  = ws + off; off += (size_t)M_ROWS * DI_;       // delta, then gated y
    float* ysb  = ws + off; off += (size_t)M_ROWS * DI_;
    _Float16* w16 = (_Float16*)(ws + off);                     // f16 Wt (<=0.82MB)

    auto wcast = [&](const float* W, int K, int N) {
        const int tot = K * N;
        wcast_kernel<<<dim3((tot + 255) / 256), dim3(256), 0, stream>>>(W, w16, K, N);
    };
    auto gemm = [&](const float* A, int K, int N,
                    const float* bias, const float* resid, float* C) {
        dim3 g(M_ROWS / 64, (N + 63) / 64);
        gemm_wmma_kernel<<<g, dim3(128), 0, stream>>>(A, w16, bias, resid, C, K, N);
    };
    auto layernorm = [&](const float* X, const float* s, const float* b, float* Y) {
        layernorm_kernel<<<dim3(M_ROWS / 8), dim3(256), 0, stream>>>(X, s, b, Y, DM);
    };

    const int NE = M_ROWS * DI_;
    const dim3 eg((NE + 255) / 256), eb(256);

    // ---- input projection: h = x @ W + b ----
    wcast(in(1), DIN, DM);
    gemm(x_in, DIN, DM, in(2), nullptr, hbuf);

    // ---- attention layer helper ----
    auto attn_layer = [&](int base) {
        layernorm(hbuf, in(base + 0), in(base + 1), hn);
        float* qb = t0;
        float* kb = t0 + (size_t)M_ROWS * DM;
        float* vb = t0 + (size_t)2 * M_ROWS * DM;
        float* ob = t0 + (size_t)3 * M_ROWS * DM;
        wcast(in(base + 2), DM, DM); gemm(hn, DM, DM, in(base + 6), nullptr, qb);
        wcast(in(base + 3), DM, DM); gemm(hn, DM, DM, in(base + 7), nullptr, kb);
        wcast(in(base + 4), DM, DM); gemm(hn, DM, DM, in(base + 8), nullptr, vb);
        attn_kernel<<<dim3(B_SZ * SEQ * NHEADS / 8), dim3(256), 0, stream>>>(qb, kb, vb, ob);
        wcast(in(base + 5), DM, DM);
        gemm(ob, DM, DM, in(base + 9), hbuf, hbuf);   // h += o@wo + bo
    };

    // ---- mamba layer helper ----
    auto mamba_layer = [&](int base) {
        layernorm(hbuf, in(base + 0), in(base + 1), hn);
        wcast(in(base + 2), DM, 2 * DI_);                     // in_w
        gemm(hn, DM, 2 * DI_, nullptr, nullptr, t0);          // [u|z]
        conv_silu_kernel<<<eg, eb, 0, stream>>>(t0, in(base + 3), in(base + 4), ucb);
        wcast(in(base + 5), DI_, DTR + 2 * DS);               // xproj_w
        gemm(ucb, DI_, DTR + 2 * DS, nullptr, nullptr, t1);   // xdbl
        dt_kernel<<<eg, eb, 0, stream>>>(t1, in(base + 6), in(base + 7), dlt);
        scan_kernel<<<dim3(B_SZ * (DI_ / 16)), dim3(256), 0, stream>>>(
            dlt, ucb, t1, in(base + 8), ysb);
        gate_kernel<<<eg, eb, 0, stream>>>(ysb, ucb, in(base + 9), t0, dlt);
        wcast(in(base + 10), DI_, DM);                        // out_w
        gemm(dlt, DI_, DM, nullptr, hbuf, hbuf);              // h += y@out_w
    };

    for (int i = 0; i < 2; ++i) attn_layer(FRONT0 + i * 10);
    for (int i = 0; i < 8; ++i) mamba_layer(TRUNK0 + i * 11);
    for (int i = 0; i < 2; ++i) attn_layer(BACK0 + i * 10);

    // ---- output head + L2 normalize ----
    wcast(in(HEADW), DM, DIN);
    gemm(hbuf, DM, DIN, nullptr, nullptr, t0);                // [8192 x 768]
    outnorm_kernel<<<dim3(M_ROWS / 8), dim3(256), 0, stream>>>(t0, (float*)d_out);
}
